// SeqCoResModel_25220047962561
// MI455X (gfx1250) — matrix operand order repaired and checked by reading, not compiled, not run on hardware
//
#include <hip/hip_runtime.h>
#include <hip/hip_bf16.h>
#include <math.h>

typedef __bf16 bf16_t;
typedef __attribute__((ext_vector_type(16))) __bf16 v16bf;
typedef __attribute__((ext_vector_type(8)))  __bf16 v8bf;
typedef __attribute__((ext_vector_type(8)))  float  v8f;

#define NSTEPS 8
#define BB 1024
#define CC 512
#define DD 256
#define KK 8192
#define HD 512
#define DECAYF 0.99f
#define EPSF 1e-5f
#define COMMITF 0.25f

// ---------------------------------------------------------------------------
// WMMA fragment helpers (CDNA5 16x16x32 bf16, wave32 layouts per ISA 7.12.2)
// ---------------------------------------------------------------------------
__device__ __forceinline__ v16bf load_a_frag(const bf16_t* __restrict__ A,
                                             int lda, int m0, int k0, int lane) {
  // A is [M,K] row-major. Lane l (m = l&15, hi = l>>4) holds row m0+m,
  // K runs [k0+8*hi .. +7] and [k0+16+8*hi .. +7].
  int m  = m0 + (lane & 15);
  int hi = (lane >> 4) & 1;
  const bf16_t* p = A + (size_t)m * lda + k0 + 8 * hi;
  v8bf lo = *(const v8bf*)(p);
  v8bf hg = *(const v8bf*)(p + 16);
  v16bf r;
#pragma unroll
  for (int i = 0; i < 8; ++i) { r[i] = lo[i]; r[8 + i] = hg[i]; }
  return r;
}

// Same A-fragment gather, but from an LDS-resident 16xDD strip (row 0 = strip
// base). Inlined into the kernel so addrspace inference emits ds_load_b128.
__device__ __forceinline__ v16bf load_a_frag_lds(const bf16_t* zs, int k0, int lane) {
  int m  = lane & 15;
  int hi = (lane >> 4) & 1;
  const bf16_t* p = zs + m * DD + k0 + 8 * hi;
  v8bf lo = *(const v8bf*)(p);
  v8bf hg = *(const v8bf*)(p + 16);
  v16bf r;
#pragma unroll
  for (int i = 0; i < 8; ++i) { r[i] = lo[i]; r[8 + i] = hg[i]; }
  return r;
}

__device__ __forceinline__ v16bf load_b_frag(const bf16_t* __restrict__ W,
                                             int ldb, int n0, int k0, int lane) {
  // W is [N,K] row-major (weights). Lane l (n = l&15, hi = l>>4) holds
  // column n0+n of B = Wᵀ, K runs [k0+8*hi .. +7] and [k0+16+8*hi .. +7].
  int n  = n0 + (lane & 15);
  int hi = (lane >> 4) & 1;
  const bf16_t* p = W + (size_t)n * ldb + k0 + 8 * hi;
  v8bf lo = *(const v8bf*)(p);
  v8bf hg = *(const v8bf*)(p + 16);
  v16bf r;
#pragma unroll
  for (int i = 0; i < 8; ++i) { r[i] = lo[i]; r[8 + i] = hg[i]; }
  return r;
}

__device__ __forceinline__ v8f wmma_bf16(v16bf a, v16bf b, v8f c) {
  return __builtin_amdgcn_wmma_f32_16x16x32_bf16(
      /*neg_a=*/false, a, /*neg_b=*/false, b,
      /*c_mod=*/(short)0, c, /*reuse_a=*/false, /*reuse_b=*/false);
}

// ---------------------------------------------------------------------------
// Generic GEMM: C[M,N] = A[M,K] @ W[N,K]^T + bias, optional ReLU,
// optional f32 and/or bf16 outputs. Block = 256 thr (8 waves), wave grid 2x4,
// wave tile 32x32 -> WG tile 64x128.
// ---------------------------------------------------------------------------
__global__ __launch_bounds__(256) void gemm_bf16_wmma(
    const bf16_t* __restrict__ A, const bf16_t* __restrict__ W,
    const float* __restrict__ bias, float* __restrict__ Cf,
    bf16_t* __restrict__ Cb, int M, int N, int K, int do_relu) {
  const int lane = threadIdx.x & 31;
  const int wave = threadIdx.x >> 5;
  const int wm = wave >> 2;           // 0..1
  const int wn = wave & 3;            // 0..3
  const int m0 = blockIdx.x * 64 + wm * 32;
  const int n0 = blockIdx.y * 128 + wn * 32;

  v8f acc00 = {}, acc01 = {}, acc10 = {}, acc11 = {};
  for (int k0 = 0; k0 < K; k0 += 32) {
    v16bf a0 = load_a_frag(A, K, m0,      k0, lane);
    v16bf a1 = load_a_frag(A, K, m0 + 16, k0, lane);
    v16bf b0 = load_b_frag(W, K, n0,      k0, lane);
    v16bf b1 = load_b_frag(W, K, n0 + 16, k0, lane);
    if (k0 + 32 < K)  // pull next B tile toward WGP$ / L2
      __builtin_prefetch(W + (size_t)(n0 + (lane & 15)) * K + k0 + 32, 0, 1);
    acc00 = wmma_bf16(a0, b0, acc00);
    acc01 = wmma_bf16(a0, b1, acc01);
    acc10 = wmma_bf16(a1, b0, acc10);
    acc11 = wmma_bf16(a1, b1, acc11);
  }

  const int ln = lane & 15, hi = (lane >> 4) & 1;
#pragma unroll
  for (int tm = 0; tm < 2; ++tm) {
#pragma unroll
    for (int tn = 0; tn < 2; ++tn) {
      v8f acc = (tm == 0) ? (tn == 0 ? acc00 : acc01)
                          : (tn == 0 ? acc10 : acc11);
      int col = n0 + tn * 16 + ln;
      float bv = bias ? bias[col] : 0.0f;
#pragma unroll
      for (int i = 0; i < 8; ++i) {
        int row = m0 + tm * 16 + i + 8 * hi;
        float v = acc[i] + bv;
        if (do_relu) v = fmaxf(v, 0.0f);
        size_t o = (size_t)row * N + col;
        if (Cf) Cf[o] = v;
        if (Cb) Cb[o] = (bf16_t)v;
      }
    }
  }
}

// ---------------------------------------------------------------------------
// VQ scoring: per WG (256 thr) a 16-row strip x 1024-col K-chunk of
// score = ||cb||^2 - 2 z.cb. The loop-invariant z strip (16x256 bf16 = 8KB)
// is staged into LDS once via GLOBAL_LOAD_ASYNC_TO_LDS_B128 (ASYNCcnt path),
// so the vmem pipe is left exclusively to codebook streaming; A fragments are
// then served from LDS. Fused WMMA + wave32 shuffle argmin reduction.
// Grid: (B/16, K/1024).
// ---------------------------------------------------------------------------
__global__ __launch_bounds__(256) void vq_score_kernel(
    const bf16_t* __restrict__ Zb, const bf16_t* __restrict__ CB,
    const float* __restrict__ cbn, float* __restrict__ ps,
    int* __restrict__ pi) {
  const int lane = threadIdx.x & 31;
  const int wave = threadIdx.x >> 5;
  const int m0 = blockIdx.x * 16;
  const int chunk = blockIdx.y;
  const int nb = chunk * 1024 + wave * 128;

  __shared__ __align__(16) bf16_t zs[16 * DD];   // 8 KB z strip

  // --- async global->LDS staging of the z strip (512 x 16B transfers) ---
  {
    const char* gsrc = (const char*)(Zb + (size_t)m0 * DD);
    unsigned lbase = (unsigned)(uintptr_t)(&zs[0]);
#pragma unroll
    for (int it = 0; it < 2; ++it) {
      unsigned byte_off = (unsigned)(threadIdx.x + it * 256) * 16u;
      unsigned laddr = lbase + byte_off;
      const char* gaddr = gsrc + byte_off;
      asm volatile("global_load_async_to_lds_b128 %0, %1, off"
                   :: "v"(laddr), "v"(gaddr)
                   : "memory");
    }
    asm volatile("s_wait_asynccnt 0x0" ::: "memory");
  }
  __syncthreads();

  v8f acc[8] = {};
  for (int k0 = 0; k0 < DD; k0 += 32) {
    v16bf a = load_a_frag_lds(zs, k0, lane);
#pragma unroll
    for (int nt = 0; nt < 8; ++nt) {
      v16bf b = load_b_frag(CB, DD, nb + nt * 16, k0, lane);
      acc[nt] = wmma_bf16(a, b, acc[nt]);
    }
  }

  const int ln = lane & 15, hi = (lane >> 4) & 1;
  float best[8];
  int bidx[8];
#pragma unroll
  for (int i = 0; i < 8; ++i) { best[i] = 3.4e38f; bidx[i] = 0; }
#pragma unroll
  for (int nt = 0; nt < 8; ++nt) {
    int col = nb + nt * 16 + ln;
    float cn = cbn[col];
#pragma unroll
    for (int i = 0; i < 8; ++i) {
      float s = cn - 2.0f * acc[nt][i];
      if (s < best[i]) { best[i] = s; bidx[i] = col; }
    }
  }

  __shared__ float ss[8][16];
  __shared__ int   si[8][16];
#pragma unroll
  for (int i = 0; i < 8; ++i) {
    float s = best[i];
    int bi = bidx[i];
#pragma unroll
    for (int off = 8; off > 0; off >>= 1) {   // stays within 16-lane half
      float s2 = __shfl_xor(s, off);
      int   i2 = __shfl_xor(bi, off);
      if (s2 < s || (s2 == s && i2 < bi)) { s = s2; bi = i2; }
    }
    if (ln == 0) {
      int r = i + 8 * hi;   // row within strip
      ss[wave][r] = s;
      si[wave][r] = bi;
    }
  }
  __syncthreads();
  if (threadIdx.x < 16) {
    float s = ss[0][threadIdx.x];
    int bi = si[0][threadIdx.x];
    for (int w = 1; w < 8; ++w) {
      float s2 = ss[w][threadIdx.x];
      int i2 = si[w][threadIdx.x];
      if (s2 < s || (s2 == s && i2 < bi)) { s = s2; bi = i2; }
    }
    int row = m0 + threadIdx.x;
    ps[row * 8 + chunk] = s;
    pi[row * 8 + chunk] = bi;
  }
}

// ---------------------------------------------------------------------------
// VQ finalize: per-row argmin over chunks, gather z_q (OLD codebook), emit
// outputs, EMA scatter (counts/dw) and commit-loss accumulation.
// ---------------------------------------------------------------------------
__global__ __launch_bounds__(64) void vq_finalize_kernel(
    const float* __restrict__ ps, const int* __restrict__ pi,
    const float* __restrict__ z, const float* __restrict__ cb,
    float* __restrict__ codes_out, float* __restrict__ idx_out,
    float* __restrict__ zc_out, bf16_t* __restrict__ prev_b,
    float* __restrict__ counts, float* __restrict__ dw,
    float* __restrict__ vq_acc, int t) {
  __shared__ int sk;
  int b = blockIdx.x;
  if (threadIdx.x == 0) {
    float s = ps[b * 8];
    int bi = pi[b * 8];
    for (int c = 1; c < 8; ++c) {
      float s2 = ps[b * 8 + c];
      int i2 = pi[b * 8 + c];
      if (s2 < s || (s2 == s && i2 < bi)) { s = s2; bi = i2; }
    }
    sk = bi;
    idx_out[b * NSTEPS + t] = (float)bi;
    atomicAdd(&counts[bi], 1.0f);
  }
  __syncthreads();
  int k = sk;
  float lsum = 0.0f;
  for (int d = threadIdx.x; d < DD; d += 64) {
    float zq = cb[(size_t)k * DD + d];
    float zv = z[(size_t)b * DD + d];
    size_t o = ((size_t)b * NSTEPS + t) * DD + d;
    codes_out[o] = zq;           // z_q_ste == z_q in value
    zc_out[o]    = zv;
    prev_b[(size_t)b * DD + d] = (bf16_t)zq;
    atomicAdd(&dw[(size_t)k * DD + d], zv);
    float df = zv - zq;
    lsum += df * df;
  }
  for (int off = 16; off > 0; off >>= 1) lsum += __shfl_xor(lsum, off);
  __shared__ float sw[2];
  if ((threadIdx.x & 31) == 0) sw[threadIdx.x >> 5] = lsum;
  __syncthreads();
  if (threadIdx.x == 0) atomicAdd(vq_acc, sw[0] + sw[1]);
}

// ---------------------------------------------------------------------------
// Elementwise / reduction glue kernels
// ---------------------------------------------------------------------------
__global__ void gru_combine_kernel(const float* __restrict__ gx,
                                   const float* __restrict__ gh,
                                   float* __restrict__ h,
                                   bf16_t* __restrict__ hb) {
  int idx = blockIdx.x * blockDim.x + threadIdx.x;
  if (idx >= BB * HD) return;
  int b = idx >> 9, j = idx & 511;
  const float* gxr = gx + (size_t)b * 3 * HD;
  const float* ghr = gh + (size_t)b * 3 * HD;
  float r  = 1.0f / (1.0f + __expf(-(gxr[j] + ghr[j])));
  float zg = 1.0f / (1.0f + __expf(-(gxr[HD + j] + ghr[HD + j])));
  float n  = tanhf(gxr[2 * HD + j] + r * ghr[2 * HD + j]);
  float hv = h[idx];
  float hn = (1.0f - zg) * n + zg * hv;
  h[idx] = hn;
  hb[idx] = (bf16_t)hn;
}

__global__ void film_kernel(const float* __restrict__ gbuf,
                            const float* __restrict__ sm,
                            const bf16_t* __restrict__ hb,
                            bf16_t* __restrict__ pin) {
  int idx = blockIdx.x * blockDim.x + threadIdx.x;
  if (idx >= BB * CC) return;
  int b = idx >> 9, j = idx & 511;
  float gamma = gbuf[(size_t)b * 1024 + j];
  float beta  = gbuf[(size_t)b * 1024 + CC + j];
  float c = (1.0f + gamma) * sm[idx] + beta;
  pin[(size_t)b * 1024 + j]      = hb[idx];
  pin[(size_t)b * 1024 + CC + j] = (bf16_t)c;
}

__global__ void spatial_mean_kernel(const float* __restrict__ sf,
                                    float* __restrict__ sm) {
  int idx = blockIdx.x * blockDim.x + threadIdx.x;
  if (idx >= BB * CC) return;
  const float* p = sf + (size_t)idx * 49;
  float s = 0.0f;
  for (int i = 0; i < 49; ++i) s += p[i];
  sm[idx] = s * (1.0f / 49.0f);
}

__global__ void cvt_kernel(const float* __restrict__ s,
                           bf16_t* __restrict__ d, int n) {
  int i = blockIdx.x * blockDim.x + threadIdx.x;
  if (i < n) d[i] = (bf16_t)s[i];
}

__global__ void zero_kernel(float* __restrict__ p, int n) {
  int i = blockIdx.x * blockDim.x + threadIdx.x;
  if (i < n) p[i] = 0.0f;
}

__global__ void init_h_kernel(float* __restrict__ h, bf16_t* __restrict__ hb) {
  int i = blockIdx.x * blockDim.x + threadIdx.x;
  if (i < BB * HD) { h[i] = 0.0f; hb[i] = (bf16_t)0.0f; }
}

__global__ void init_prev_kernel(const float* __restrict__ bos,
                                 bf16_t* __restrict__ pb) {
  int i = blockIdx.x * blockDim.x + threadIdx.x;
  if (i < BB * DD) pb[i] = (bf16_t)bos[i & (DD - 1)];
}

__global__ void cb_norm_kernel(const float* __restrict__ cbf,
                               float* __restrict__ cbn) {
  int k = blockIdx.x * blockDim.x + threadIdx.x;
  if (k >= KK) return;
  const float* p = cbf + (size_t)k * DD;
  float s = 0.0f;
  for (int d = 0; d < DD; ++d) { float v = p[d]; s += v * v; }
  cbn[k] = s;
}

__global__ __launch_bounds__(256) void cs_update_kernel(
    float* __restrict__ cs, const float* __restrict__ counts,
    float* __restrict__ n_tot) {
  int k = blockIdx.x * 256 + threadIdx.x;
  float v = 0.0f;
  if (k < KK) {
    v = DECAYF * cs[k] + (1.0f - DECAYF) * counts[k];
    cs[k] = v;
  }
  for (int off = 16; off > 0; off >>= 1) v += __shfl_xor(v, off);
  __shared__ float sw[8];
  if ((threadIdx.x & 31) == 0) sw[threadIdx.x >> 5] = v;
  __syncthreads();
  if (threadIdx.x == 0) {
    float s = 0.0f;
    for (int i = 0; i < 8; ++i) s += sw[i];
    atomicAdd(n_tot, s);
  }
}

__global__ void cs_smooth_kernel(float* __restrict__ cs,
                                 const float* __restrict__ n_tot) {
  int k = blockIdx.x * blockDim.x + threadIdx.x;
  if (k >= KK) return;
  float nt = *n_tot;
  cs[k] = (cs[k] + EPSF) / (nt + (float)KK * EPSF) * nt;
}

__global__ void ew_cb_kernel(float* __restrict__ ew,
                             const float* __restrict__ dw,
                             const float* __restrict__ cs,
                             float* __restrict__ cbf,
                             bf16_t* __restrict__ cbb) {
  int i = blockIdx.x * blockDim.x + threadIdx.x;
  if (i >= KK * DD) return;
  int k = i >> 8;
  float e = DECAYF * ew[i] + (1.0f - DECAYF) * dw[i];
  ew[i] = e;
  float c = e / cs[k];
  cbf[i] = c;
  cbb[i] = (bf16_t)c;
}

__global__ void loss_final_kernel(const float* __restrict__ vq_acc,
                                  float* __restrict__ out) {
  if (blockIdx.x == 0 && threadIdx.x == 0)
    out[0] = COMMITF * vq_acc[0] / ((float)BB * (float)DD * (float)NSTEPS);
}

// ---------------------------------------------------------------------------
// Host orchestration
// ---------------------------------------------------------------------------
extern "C" void kernel_launch(void* const* d_in, const int* in_sizes, int n_in,
                              void* d_out, int out_size, void* d_ws,
                              size_t ws_size, hipStream_t stream) {
  (void)in_sizes; (void)n_in; (void)out_size; (void)ws_size;
  const float* sf   = (const float*)d_in[0];   // [1024,512,7,7]
  const float* bos  = (const float*)d_in[1];   // [256]
  const float* W_ih = (const float*)d_in[2];   // [1536,256]
  const float* W_hh = (const float*)d_in[3];   // [1536,512]
  const float* b_ih = (const float*)d_in[4];
  const float* b_hh = (const float*)d_in[5];
  const float* Wg   = (const float*)d_in[6];   // [512,512]
  const float* bg   = (const float*)d_in[7];
  const float* Wb   = (const float*)d_in[8];   // [512,512]
  const float* bb   = (const float*)d_in[9];
  const float* W1   = (const float*)d_in[10];  // [512,1024]
  const float* b1   = (const float*)d_in[11];
  const float* W2   = (const float*)d_in[12];  // [256,512]
  const float* b2   = (const float*)d_in[13];
  const float* codebook = (const float*)d_in[14]; // [8192,256]
  const float* ema_cs   = (const float*)d_in[15]; // [8192]
  const float* ema_w    = (const float*)d_in[16]; // [8192,256]

  // --- workspace bump allocator (256B aligned) ---
  char* wp = (char*)d_ws;
  auto alloc = [&](size_t bytes) -> char* {
    char* p = wp;
    wp += (bytes + 255) & ~(size_t)255;
    return p;
  };
  float*  cb_f    = (float*)alloc((size_t)KK * DD * 4);
  bf16_t* cb_b    = (bf16_t*)alloc((size_t)KK * DD * 2);
  float*  cb_norm = (float*)alloc(KK * 4);
  float*  cs      = (float*)alloc(KK * 4);
  float*  ew      = (float*)alloc((size_t)KK * DD * 4);
  // counts | dw | n_tot contiguous so one zero_kernel clears all three
  float*  counts  = (float*)alloc(KK * 4 + (size_t)KK * DD * 4 + 256);
  float*  dw      = counts + KK;
  float*  n_tot   = dw + (size_t)KK * DD;
  float*  vq_acc  = (float*)alloc(256);
  float*  h       = (float*)alloc((size_t)BB * HD * 4);
  bf16_t* h_b     = (bf16_t*)alloc((size_t)BB * HD * 2);
  bf16_t* prev_b  = (bf16_t*)alloc((size_t)BB * DD * 2);
  float*  gx      = (float*)alloc((size_t)BB * 3 * HD * 4);
  float*  gh      = (float*)alloc((size_t)BB * 3 * HD * 4);
  float*  gbuf    = (float*)alloc((size_t)BB * 1024 * 4);
  bf16_t* pin_b   = (bf16_t*)alloc((size_t)BB * 1024 * 2);
  bf16_t* t1_b    = (bf16_t*)alloc((size_t)BB * HD * 2);
  float*  z       = (float*)alloc((size_t)BB * DD * 4);
  bf16_t* z_b     = (bf16_t*)alloc((size_t)BB * DD * 2);
  float*  sm      = (float*)alloc((size_t)BB * CC * 4);
  float*  ps      = (float*)alloc(BB * 8 * 4);
  int*    pi      = (int*)alloc(BB * 8 * 4);
  bf16_t* Wih_b   = (bf16_t*)alloc((size_t)3 * HD * DD * 2);
  bf16_t* Whh_b   = (bf16_t*)alloc((size_t)3 * HD * HD * 2);
  bf16_t* Wgb_b   = (bf16_t*)alloc((size_t)1024 * HD * 2);
  bf16_t* W1_b    = (bf16_t*)alloc((size_t)HD * 1024 * 2);
  bf16_t* W2_b    = (bf16_t*)alloc((size_t)DD * HD * 2);
  float*  bgbb    = (float*)alloc(1024 * 4);

  const int T = 256;
  auto blocks = [](size_t n, int t) { return (unsigned)((n + t - 1) / t); };

  // --- one-time setup ---
  hipMemcpyAsync(cb_f, codebook, (size_t)KK * DD * 4, hipMemcpyDeviceToDevice, stream);
  hipMemcpyAsync(cs, ema_cs, KK * 4, hipMemcpyDeviceToDevice, stream);
  hipMemcpyAsync(ew, ema_w, (size_t)KK * DD * 4, hipMemcpyDeviceToDevice, stream);
  hipMemcpyAsync(bgbb, bg, CC * 4, hipMemcpyDeviceToDevice, stream);
  hipMemcpyAsync(bgbb + CC, bb, CC * 4, hipMemcpyDeviceToDevice, stream);
  cvt_kernel<<<blocks((size_t)KK * DD, T), T, 0, stream>>>(codebook, cb_b, KK * DD);
  cvt_kernel<<<blocks((size_t)3 * HD * DD, T), T, 0, stream>>>(W_ih, Wih_b, 3 * HD * DD);
  cvt_kernel<<<blocks((size_t)3 * HD * HD, T), T, 0, stream>>>(W_hh, Whh_b, 3 * HD * HD);
  cvt_kernel<<<blocks((size_t)CC * HD, T), T, 0, stream>>>(Wg, Wgb_b, CC * HD);
  cvt_kernel<<<blocks((size_t)CC * HD, T), T, 0, stream>>>(Wb, Wgb_b + (size_t)CC * HD, CC * HD);
  cvt_kernel<<<blocks((size_t)HD * 1024, T), T, 0, stream>>>(W1, W1_b, HD * 1024);
  cvt_kernel<<<blocks((size_t)DD * HD, T), T, 0, stream>>>(W2, W2_b, DD * HD);
  cb_norm_kernel<<<blocks(KK, T), T, 0, stream>>>(cb_f, cb_norm);
  spatial_mean_kernel<<<blocks((size_t)BB * CC, T), T, 0, stream>>>(sf, sm);
  init_h_kernel<<<blocks((size_t)BB * HD, T), T, 0, stream>>>(h, h_b);
  init_prev_kernel<<<blocks((size_t)BB * DD, T), T, 0, stream>>>(bos, prev_b);
  zero_kernel<<<1, 1, 0, stream>>>(vq_acc, 1);

  float* out      = (float*)d_out;
  float* out_h    = out;                                 // [1024,512]
  float* out_code = out_h + (size_t)BB * HD;             // [1024,8,256]
  float* out_idx  = out_code + (size_t)BB * NSTEPS * DD; // [1024,8]
  float* out_zc   = out_idx + (size_t)BB * NSTEPS;       // [1024,8,256]
  float* out_loss = out_zc + (size_t)BB * NSTEPS * DD;   // [1]

  for (int t = 0; t < NSTEPS; ++t) {
    // GRU gate GEMMs: gx = prev@W_ih^T + b_ih ; gh = h@W_hh^T + b_hh
    gemm_bf16_wmma<<<dim3(BB / 64, (3 * HD) / 128), 256, 0, stream>>>(
        prev_b, Wih_b, b_ih, gx, nullptr, BB, 3 * HD, DD, 0);
    gemm_bf16_wmma<<<dim3(BB / 64, (3 * HD) / 128), 256, 0, stream>>>(
        h_b, Whh_b, b_hh, gh, nullptr, BB, 3 * HD, HD, 0);
    gru_combine_kernel<<<blocks((size_t)BB * HD, T), T, 0, stream>>>(gx, gh, h, h_b);

    // FiLM: [gamma|beta] = h@[Wg;Wb]^T + [bg|bb] ; pin = [h, c]
    gemm_bf16_wmma<<<dim3(BB / 64, 1024 / 128), 256, 0, stream>>>(
        h_b, Wgb_b, bgbb, gbuf, nullptr, BB, 1024, HD, 0);
    film_kernel<<<blocks((size_t)BB * CC, T), T, 0, stream>>>(gbuf, sm, h_b, pin_b);

    // MLP: t1 = relu(pin@W1^T + b1) ; z = t1@W2^T + b2
    gemm_bf16_wmma<<<dim3(BB / 64, HD / 128), 256, 0, stream>>>(
        pin_b, W1_b, b1, nullptr, t1_b, BB, HD, 1024, 1);
    gemm_bf16_wmma<<<dim3(BB / 64, DD / 128), 256, 0, stream>>>(
        t1_b, W2_b, b2, z, z_b, BB, DD, HD, 0);

    // zero counts + dw + n_tot (contiguous)
    zero_kernel<<<blocks((size_t)KK + (size_t)KK * DD + 64, T), T, 0, stream>>>(
        counts, KK + KK * DD + 64);

    // VQ nearest neighbor (WMMA + async LDS staging) + finalize
    vq_score_kernel<<<dim3(BB / 16, KK / 1024), 256, 0, stream>>>(
        z_b, cb_b, cb_norm, ps, pi);
    vq_finalize_kernel<<<BB, 64, 0, stream>>>(
        ps, pi, z, cb_f, out_code, out_idx, out_zc, prev_b, counts, dw, vq_acc, t);

    // EMA codebook update (result only consumed by the NEXT step)
    if (t + 1 < NSTEPS) {
      cs_update_kernel<<<KK / 256, 256, 0, stream>>>(cs, counts, n_tot);
      cs_smooth_kernel<<<blocks(KK, T), T, 0, stream>>>(cs, n_tot);
      ew_cb_kernel<<<blocks((size_t)KK * DD, T), T, 0, stream>>>(ew, dw, cs, cb_f, cb_b);
      cb_norm_kernel<<<blocks(KK, T), T, 0, stream>>>(cb_f, cb_norm);
    }
  }

  hipMemcpyAsync(out_h, h, (size_t)BB * HD * 4, hipMemcpyDeviceToDevice, stream);
  loss_final_kernel<<<1, 1, 0, stream>>>(vq_acc, out_loss);
}
